// SwinTransformerBlock_75857712382412
// MI455X (gfx1250) — compile-verified, matrix operand order checked by reading, not run on hardware
//
#include <hip/hip_runtime.h>
#include <hip/hip_bf16.h>
#include <math.h>

// ---------------------------------------------------------------------------
// Swin block (shifted-window attention + MLP) for MI455X / gfx1250.
// All GEMMs in bf16 via v_wmma_f32_16x16x32_bf16, fp32 accumulate.
// B=32, H=W=56, WS=7 (N=49 -> padded 64), NH=12, hd=32, DIM=384, HID=1536.
// ---------------------------------------------------------------------------

typedef __attribute__((ext_vector_type(16))) __bf16 v16bf;
typedef __attribute__((ext_vector_type(8)))  float  v8f;
typedef unsigned short u16;

union Frag { v16bf v; uint4 q[2]; };

__device__ inline u16 f2bf(float f) {
  unsigned int u = __float_as_uint(f);
  unsigned int r = u + 0x7FFFu + ((u >> 16) & 1u);
  return (u16)(r >> 16);
}

__device__ inline v8f vzero8() {
  v8f z = {0.f,0.f,0.f,0.f,0.f,0.f,0.f,0.f};
  return z;
}

__device__ inline void store_out(float* p, float v) { *p = v; }
__device__ inline void store_out(u16* p, float v)   { *p = f2bf(v); }

// ---------------------------------------------------------------------------
// fp32 -> bf16 convert (weights)
// ---------------------------------------------------------------------------
__global__ void f2bf_kernel(const float* __restrict__ in, u16* __restrict__ out, int n) {
  for (int i = blockIdx.x * blockDim.x + threadIdx.x; i < n; i += gridDim.x * blockDim.x)
    out[i] = f2bf(in[i]);
}

// ---------------------------------------------------------------------------
// Window partition with cyclic shift (-3,-3); pad token rows 49..63 with 0.
// xw: [B*64 windows][64 tokens][384] bf16
// ---------------------------------------------------------------------------
__global__ __launch_bounds__(128) void win_partition(const float* __restrict__ x,
                                                     u16* __restrict__ xw) {
  int idx  = blockIdx.x;          // winG*64 + t
  int t    = idx & 63;
  int winG = idx >> 6;
  int b    = winG >> 6;
  int win  = winG & 63;
  u16* dst = xw + ((size_t)winG * 64 + t) * 384;
  if (t >= 49) {
    for (int c = threadIdx.x; c < 384; c += 128) dst[c] = 0;
    return;
  }
  int wi = win >> 3, wj = win & 7;
  int ti = t / 7,   tj = t % 7;
  int sh = (wi * 7 + ti + 3) % 56;
  int sw = (wj * 7 + tj + 3) % 56;
  const float* src = x + ((size_t)b * 3136 + sh * 56 + sw) * 384;
  for (int c = threadIdx.x; c < 384; c += 128) dst[c] = f2bf(src[c]);
}

// ---------------------------------------------------------------------------
// comb: rel-pos bias + shift mask (+ -1e30 padding), PRE-SWIZZLED into the
// WMMA D-fragment order so the attention kernel reads contiguous b128:
//   offset = s*1024 + lane*32 + nt*8 + r
//   maps to n = s*16 + r + (lane>>4)*8 ,  m = nt*16 + (lane&15)
// ---------------------------------------------------------------------------
__global__ __launch_bounds__(256) void build_comb(const float* __restrict__ rpb,
                                                  const float* __restrict__ mask,
                                                  const int* __restrict__ rel,
                                                  float* __restrict__ comb) {
  int h   = blockIdx.x % 12;
  int wIm = blockIdx.x / 12;
  float* dst = comb + (size_t)blockIdx.x * 4096;
  for (int e = threadIdx.x; e < 4096; e += 256) {
    int s    = e >> 10;
    int lane = (e >> 5) & 31;
    int nt   = (e >> 3) & 3;
    int r    = e & 7;
    int n = s * 16 + r + (lane >> 4) * 8;
    int m = nt * 16 + (lane & 15);
    float v;
    if (m >= 49)      v = -1e30f;
    else if (n >= 49) v = 0.0f;
    else v = rpb[rel[n * 49 + m] * 12 + h] + mask[((size_t)wIm * 49 + n) * 49 + m];
    dst[e] = v;
  }
}

// ---------------------------------------------------------------------------
// Generic bf16 WMMA GEMM:  C[M,N] = A[M,K] * W[N,K]^T + bias (+GELU)
// block 256 (8 waves), tile 128x64, K-step 64 (8 WMMA per barrier pair/wave).
// ---------------------------------------------------------------------------
#define LDA 72   // LDS row stride in bf16 elems: 144B (16B aligned, conflict-free)

template<int GELU, typename OutT>
__global__ __launch_bounds__(256) void gemm_bf16(const u16* __restrict__ A,
                                                 const u16* __restrict__ W,
                                                 const float* __restrict__ bias,
                                                 OutT* __restrict__ C,
                                                 int M, int N, int K) {
  __shared__ u16 sA[128 * LDA];
  __shared__ u16 sB[64 * LDA];
  const int tid  = threadIdx.x;
  const int lane = tid & 31;
  const int wave = tid >> 5;
  const int wm   = wave & 3;   // 32-row slab
  const int wn   = wave >> 2;  // 32-col slab
  const int m0   = blockIdx.y * 128;
  const int n0   = blockIdx.x * 64;
  const int lo   = lane & 15;
  const int hi   = lane >> 4;
  const int kb   = hi * 8;

  v8f acc[2][2];
  acc[0][0] = vzero8(); acc[0][1] = vzero8();
  acc[1][0] = vzero8(); acc[1][1] = vzero8();

  for (int k0 = 0; k0 < K; k0 += 64) {
    __syncthreads();
    // A tile 128x64 : 1024 chunks of 8 bf16 (16B)
    #pragma unroll
    for (int c = 0; c < 4; ++c) {
      int chunk = tid + c * 256;
      int row   = chunk >> 3;
      int kc    = (chunk & 7) * 8;
      *(uint4*)&sA[row * LDA + kc] =
          *(const uint4*)&A[(size_t)(m0 + row) * K + k0 + kc];
    }
    // B tile 64x64 : 512 chunks
    #pragma unroll
    for (int c = 0; c < 2; ++c) {
      int chunk = tid + c * 256;
      int row   = chunk >> 3;
      int kc    = (chunk & 7) * 8;
      *(uint4*)&sB[row * LDA + kc] =
          *(const uint4*)&W[(size_t)(n0 + row) * K + k0 + kc];
    }
    __syncthreads();

    #pragma unroll
    for (int ks = 0; ks < 2; ++ks) {
      const int kbb = ks * 32 + kb;
      Frag af[2], bfr[2];
      #pragma unroll
      for (int mt = 0; mt < 2; ++mt) {
        int r = wm * 32 + mt * 16 + lo;
        af[mt].q[0] = *(const uint4*)&sA[r * LDA + kbb];
        af[mt].q[1] = *(const uint4*)&sA[r * LDA + kbb + 16];
      }
      #pragma unroll
      for (int nt = 0; nt < 2; ++nt) {
        int r = wn * 32 + nt * 16 + lo;
        bfr[nt].q[0] = *(const uint4*)&sB[r * LDA + kbb];
        bfr[nt].q[1] = *(const uint4*)&sB[r * LDA + kbb + 16];
      }
      #pragma unroll
      for (int mt = 0; mt < 2; ++mt)
        #pragma unroll
        for (int nt = 0; nt < 2; ++nt)
          acc[mt][nt] = __builtin_amdgcn_wmma_f32_16x16x32_bf16(
              false, af[mt].v, false, bfr[nt].v, (short)0, acc[mt][nt], false, false);
    }
  }

  #pragma unroll
  for (int mt = 0; mt < 2; ++mt)
    #pragma unroll
    for (int nt = 0; nt < 2; ++nt) {
      int col = n0 + wn * 32 + nt * 16 + lo;
      float bv = bias[col];
      #pragma unroll
      for (int r = 0; r < 8; ++r) {
        int row = m0 + wm * 32 + mt * 16 + r + hi * 8;
        float v = acc[mt][nt][r] + bv;
        if (GELU) v = 0.5f * v * (1.0f + erff(v * 0.70710678118654752f));
        store_out(&C[(size_t)row * N + col], v);
      }
    }
}

// ---------------------------------------------------------------------------
// v transpose: vt[winG][h][d(32)][m(64)] from qkv[winG][m][768 + h*32 + d]
// ---------------------------------------------------------------------------
__global__ __launch_bounds__(256) void transpose_v(const u16* __restrict__ qkv,
                                                   u16* __restrict__ vt) {
  int h      = blockIdx.x % 12;
  size_t wg  = blockIdx.x / 12;
  const u16* src = qkv + wg * 64 * 1152 + 768 + h * 32;
  u16* dst = vt + (size_t)blockIdx.x * 2048;
  for (int e = threadIdx.x; e < 2048; e += 256) {
    int d = e >> 6, m = e & 63;
    dst[d * 64 + m] = src[(size_t)m * 1152 + d];
  }
}

// ---------------------------------------------------------------------------
// Attention: one wave per (window, head). S = q*k^T (scale) + comb, softmax,
// O = P*V. All matmuls via WMMA bf16; comb read as fragment-ordered b128.
// ---------------------------------------------------------------------------
__global__ __launch_bounds__(128) void attention_kernel(const u16* __restrict__ qkv,
                                                        const u16* __restrict__ vt,
                                                        const float* __restrict__ comb,
                                                        u16* __restrict__ ao) {
  __shared__ u16 sP[4][16 * 64];
  const int wave = threadIdx.x >> 5;
  const int lane = threadIdx.x & 31;
  const int lo = lane & 15, hi = lane >> 4, kb = hi * 8;
  const int winG = blockIdx.x;
  const int h    = blockIdx.y * 4 + wave;
  const int wIm  = winG & 63;

  const u16* Qb = qkv + (size_t)winG * 64 * 1152 + h * 32;
  const u16* Kb = Qb + 384;
  const u16* Vt = vt + ((size_t)winG * 12 + h) * 2048;
  const float* Cb = comb + ((size_t)wIm * 12 + h) * 4096 + lane * 32;
  u16* sp = sP[wave];

  Frag kf[4], vf[2][2];
  #pragma unroll
  for (int nt = 0; nt < 4; ++nt) {
    int m = nt * 16 + lo;
    kf[nt].q[0] = *(const uint4*)&Kb[(size_t)m * 1152 + kb];
    kf[nt].q[1] = *(const uint4*)&Kb[(size_t)m * 1152 + kb + 16];
  }
  #pragma unroll
  for (int kt = 0; kt < 2; ++kt)
    #pragma unroll
    for (int nt = 0; nt < 2; ++nt) {
      int d = nt * 16 + lo;
      vf[kt][nt].q[0] = *(const uint4*)&Vt[d * 64 + kt * 32 + kb];
      vf[kt][nt].q[1] = *(const uint4*)&Vt[d * 64 + kt * 32 + kb + 16];
    }

  const float scale = 0.17677669529663687f; // 1/sqrt(32)

  for (int s = 0; s < 4; ++s) {            // 16-row slice of the 64 tokens
    Frag qf;
    int n = s * 16 + lo;
    qf.q[0] = *(const uint4*)&Qb[(size_t)n * 1152 + kb];
    qf.q[1] = *(const uint4*)&Qb[(size_t)n * 1152 + kb + 16];

    v8f sacc[4];
    #pragma unroll
    for (int nt = 0; nt < 4; ++nt)
      sacc[nt] = __builtin_amdgcn_wmma_f32_16x16x32_bf16(
          false, qf.v, false, kf[nt].v, (short)0, vzero8(), false, false);

    // fragment-ordered comb: 8 contiguous b128 per lane
    float4 cq[8];
    const float4* Cs = (const float4*)(Cb + (s << 10));
    #pragma unroll
    for (int i = 0; i < 8; ++i) cq[i] = Cs[i];
    const float* cv = (const float*)cq;

    float sv[4][8], rmax[8], rsum[8];
    #pragma unroll
    for (int r = 0; r < 8; ++r) {
      float mx = -3.0e38f;
      #pragma unroll
      for (int nt = 0; nt < 4; ++nt) {
        float t = sacc[nt][r] * scale + cv[nt * 8 + r];
        sv[nt][r] = t;
        mx = fmaxf(mx, t);
      }
      rmax[r] = mx;
    }
    #pragma unroll
    for (int d = 1; d < 16; d <<= 1)
      #pragma unroll
      for (int r = 0; r < 8; ++r)
        rmax[r] = fmaxf(rmax[r], __shfl_xor(rmax[r], d, 32));
    #pragma unroll
    for (int r = 0; r < 8; ++r) {
      float acc = 0.f;
      #pragma unroll
      for (int nt = 0; nt < 4; ++nt) {
        float e = __expf(sv[nt][r] - rmax[r]);
        sv[nt][r] = e;
        acc += e;
      }
      rsum[r] = acc;
    }
    #pragma unroll
    for (int d = 1; d < 16; d <<= 1)
      #pragma unroll
      for (int r = 0; r < 8; ++r)
        rsum[r] += __shfl_xor(rsum[r], d, 32);
    #pragma unroll
    for (int r = 0; r < 8; ++r) rsum[r] = 1.0f / rsum[r];

    // P -> LDS (D-layout) then re-fragment as A-layout for P*V
    #pragma unroll
    for (int r = 0; r < 8; ++r)
      #pragma unroll
      for (int nt = 0; nt < 4; ++nt)
        sp[(r + hi * 8) * 64 + nt * 16 + lo] = f2bf(sv[nt][r] * rsum[r]);
    asm volatile("s_wait_dscnt 0" ::: "memory");

    Frag pf[2];
    #pragma unroll
    for (int kt = 0; kt < 2; ++kt) {
      pf[kt].q[0] = *(const uint4*)&sp[lo * 64 + kt * 32 + kb];
      pf[kt].q[1] = *(const uint4*)&sp[lo * 64 + kt * 32 + kb + 16];
    }
    asm volatile("s_wait_dscnt 0" ::: "memory");

    v8f oacc[2];
    oacc[0] = vzero8(); oacc[1] = vzero8();
    #pragma unroll
    for (int kt = 0; kt < 2; ++kt)
      #pragma unroll
      for (int nt = 0; nt < 2; ++nt)
        oacc[nt] = __builtin_amdgcn_wmma_f32_16x16x32_bf16(
            false, pf[kt].v, false, vf[kt][nt].v, (short)0, oacc[nt], false, false);

    #pragma unroll
    for (int nt = 0; nt < 2; ++nt)
      #pragma unroll
      for (int r = 0; r < 8; ++r) {
        int tok = s * 16 + r + hi * 8;
        ao[((size_t)winG * 64 + tok) * 384 + h * 32 + nt * 16 + lo] =
            f2bf(oacc[nt][r]);
      }
  }
}

// ---------------------------------------------------------------------------
// Window reverse + roll back + LN + residual:
//   x1 = x + LN(proj_out gathered from window layout); also x1 in bf16.
// One wave per token (384 ch = 12/lane).
// ---------------------------------------------------------------------------
__global__ __launch_bounds__(128) void ln_residual_win(const float* __restrict__ y,
                                                       const float* __restrict__ x,
                                                       const float* __restrict__ g,
                                                       const float* __restrict__ be,
                                                       float* __restrict__ x1,
                                                       u16* __restrict__ x1b) {
  const int wave = threadIdx.x >> 5;
  const int lane = threadIdx.x & 31;
  size_t tok = (size_t)blockIdx.x * 4 + wave;   // 0..100351
  int b = (int)(tok / 3136);
  int l = (int)(tok % 3136);
  int hh = l / 56, ww = l % 56;
  int sh = (hh + 53) % 56, sw = (ww + 53) % 56; // reverse roll (+3,+3)
  int winG = b * 64 + (sh / 7) * 8 + (sw / 7);
  int t = (sh % 7) * 7 + (sw % 7);
  const float* yr = y + ((size_t)winG * 64 + t) * 384;
  const float* xr = x + tok * 384;

  float v[12];
  float sum = 0.f;
  #pragma unroll
  for (int i = 0; i < 12; ++i) { v[i] = yr[lane + i * 32]; sum += v[i]; }
  #pragma unroll
  for (int d = 1; d < 32; d <<= 1) sum += __shfl_xor(sum, d, 32);
  float mean = sum * (1.0f / 384.0f);
  float vs = 0.f;
  #pragma unroll
  for (int i = 0; i < 12; ++i) { float dd = v[i] - mean; vs += dd * dd; }
  #pragma unroll
  for (int d = 1; d < 32; d <<= 1) vs += __shfl_xor(vs, d, 32);
  float rstd = rsqrtf(vs * (1.0f / 384.0f) + 1e-5f);
  #pragma unroll
  for (int i = 0; i < 12; ++i) {
    int c = lane + i * 32;
    float o = (v[i] - mean) * rstd * g[c] + be[c] + xr[c];
    x1[tok * 384 + c]  = o;
    x1b[tok * 384 + c] = f2bf(o);
  }
}

// out = x1 + LN(y2)
__global__ __launch_bounds__(128) void ln_final(const float* __restrict__ y,
                                                const float* __restrict__ x1,
                                                const float* __restrict__ g,
                                                const float* __restrict__ be,
                                                float* __restrict__ out) {
  const int wave = threadIdx.x >> 5;
  const int lane = threadIdx.x & 31;
  size_t tok = (size_t)blockIdx.x * 4 + wave;
  const float* yr = y + tok * 384;
  const float* xr = x1 + tok * 384;
  float v[12];
  float sum = 0.f;
  #pragma unroll
  for (int i = 0; i < 12; ++i) { v[i] = yr[lane + i * 32]; sum += v[i]; }
  #pragma unroll
  for (int d = 1; d < 32; d <<= 1) sum += __shfl_xor(sum, d, 32);
  float mean = sum * (1.0f / 384.0f);
  float vs = 0.f;
  #pragma unroll
  for (int i = 0; i < 12; ++i) { float dd = v[i] - mean; vs += dd * dd; }
  #pragma unroll
  for (int d = 1; d < 32; d <<= 1) vs += __shfl_xor(vs, d, 32);
  float rstd = rsqrtf(vs * (1.0f / 384.0f) + 1e-5f);
  #pragma unroll
  for (int i = 0; i < 12; ++i) {
    int c = lane + i * 32;
    out[tok * 384 + c] = (v[i] - mean) * rstd * g[c] + be[c] + xr[c];
  }
}

// ---------------------------------------------------------------------------
// Launcher. Workspace layout (time-multiplexed regions, ~886 MiB):
//   R0: Xw bf16 (100.6MB)      -> H fc1-out bf16 (308.3MB)
//   R1: qkv bf16 (302.0MB)     -> x1 fp32 (154.1MB)
//   R2: attn_out bf16 (100.7MB)-> x1 bf16 (77.1MB)
//   R3: vT bf16 (100.7MB) -> proj fp32 (201.3MB) -> y2 fp32 (154.1MB)
//   R4: comb fp32 (12.6MB);  R5: bf16 weights (3.5MB)
// ---------------------------------------------------------------------------
extern "C" void kernel_launch(void* const* d_in, const int* in_sizes, int n_in,
                              void* d_out, int out_size, void* d_ws, size_t ws_size,
                              hipStream_t stream) {
  const float* x       = (const float*)d_in[0];
  const float* qkv_w   = (const float*)d_in[1];
  const float* qkv_b   = (const float*)d_in[2];
  const float* proj_w  = (const float*)d_in[3];
  const float* proj_b  = (const float*)d_in[4];
  const float* rpb     = (const float*)d_in[5];
  const float* ln1_g   = (const float*)d_in[6];
  const float* ln1_b   = (const float*)d_in[7];
  const float* fc1_w   = (const float*)d_in[8];
  const float* fc1_b   = (const float*)d_in[9];
  const float* fc2_w   = (const float*)d_in[10];
  const float* fc2_b   = (const float*)d_in[11];
  const float* ln2_g   = (const float*)d_in[12];
  const float* ln2_b   = (const float*)d_in[13];
  const float* amask   = (const float*)d_in[14];
  const int*   rel     = (const int*)d_in[15];
  float* out = (float*)d_out;

  char* ws = (char*)d_ws;
  const size_t R0 = 0;
  const size_t R1 = R0 + 308281344ull;
  const size_t R2 = R1 + 301989888ull;
  const size_t R3 = R2 + 100663296ull;
  const size_t R4 = R3 + 201326592ull;
  const size_t R5 = R4 + 12582912ull;

  u16*   Xw    = (u16*)(ws + R0);
  u16*   Hbuf  = (u16*)(ws + R0);
  u16*   qkvb  = (u16*)(ws + R1);
  float* x1    = (float*)(ws + R1);
  u16*   ao    = (u16*)(ws + R2);
  u16*   x1b   = (u16*)(ws + R2);
  u16*   vtb   = (u16*)(ws + R3);
  float* projo = (float*)(ws + R3);
  float* y2    = (float*)(ws + R3);
  float* comb  = (float*)(ws + R4);
  u16*   wb    = (u16*)(ws + R5);
  u16* qkv_wb  = wb;
  u16* proj_wb = wb + 442368;
  u16* fc1_wb  = wb + 589824;
  u16* fc2_wb  = wb + 1179648;

  // weights -> bf16
  f2bf_kernel<<<1728, 256, 0, stream>>>(qkv_w,  qkv_wb,  442368);
  f2bf_kernel<<<576,  256, 0, stream>>>(proj_w, proj_wb, 147456);
  f2bf_kernel<<<2304, 256, 0, stream>>>(fc1_w,  fc1_wb,  589824);
  f2bf_kernel<<<2304, 256, 0, stream>>>(fc2_w,  fc2_wb,  589824);

  // shifted window partition, fragment-ordered bias+mask table
  win_partition<<<131072, 128, 0, stream>>>(x, Xw);
  build_comb<<<768, 256, 0, stream>>>(rpb, amask, rel, comb);

  // QKV: [131072,384] x [1152,384]^T
  gemm_bf16<0, u16><<<dim3(18, 1024), 256, 0, stream>>>(
      Xw, qkv_wb, qkv_b, qkvb, 131072, 1152, 384);

  transpose_v<<<24576, 256, 0, stream>>>(qkvb, vtb);

  // windowed attention (2048 windows x 12 heads)
  attention_kernel<<<dim3(2048, 3), 128, 0, stream>>>(qkvb, vtb, comb, ao);

  // proj: [131072,384] x [384,384]^T -> fp32
  gemm_bf16<0, float><<<dim3(6, 1024), 256, 0, stream>>>(
      ao, proj_wb, proj_b, projo, 131072, 384, 384);

  // window reverse + roll + LN1 + residual
  ln_residual_win<<<25088, 128, 0, stream>>>(projo, x, ln1_g, ln1_b, x1, x1b);

  // fc1 + GELU: [100352,384] x [1536,384]^T -> bf16
  gemm_bf16<1, u16><<<dim3(24, 784), 256, 0, stream>>>(
      x1b, fc1_wb, fc1_b, Hbuf, 100352, 1536, 384);

  // fc2: [100352,1536] x [384,1536]^T -> fp32
  gemm_bf16<0, float><<<dim3(6, 784), 256, 0, stream>>>(
      Hbuf, fc2_wb, fc2_b, y2, 100352, 384, 1536);

  // LN2 + residual -> out
  ln_final<<<25088, 128, 0, stream>>>(y2, x1, ln2_g, ln2_b, out);
}
